// SABlock_9929964388638
// MI455X (gfx1250) — matrix operand order repaired and checked by reading, not compiled, not run on hardware
//
#include <hip/hip_runtime.h>
#include <hip/hip_bf16.h>
#include <stdint.h>

typedef _Float16 half_t;
typedef __attribute__((ext_vector_type(16))) _Float16 v16h;
typedef __attribute__((ext_vector_type(8)))  _Float16 v8h;
typedef __attribute__((ext_vector_type(8)))  float    v8f;

#define NTOK 4096   // X*Y*Z
#define CDIM 512
#define HEADS 8
#define DHEAD 64
#define C3   1536

// LDS row pitches (halves) chosen for conflict-free b128 fragment reads:
// K tile pitch 72 -> byte stride 144 -> bank stride 36 (all 16 lanes distinct mod 64)
// V/P  pitch 40 -> byte stride  80 -> bank stride 20 (all 16 lanes distinct mod 64)
#define KPITCH 72
#define VPITCH 40
#define PPITCH 40

// ---------------- WMMA helpers (gfx1250 f16 16x16x32, fp32 acc) -------------

static __device__ __forceinline__ v8f wmma16(v16h a, v16h b, v8f c) {
    // (neg_a, A, neg_b, B, c_mod, C, reuse_a, reuse_b)
    return __builtin_amdgcn_wmma_f32_16x16x32_f16(false, a, false, b,
                                                  (short)0, c, false, false);
}

// A fragment (16x32, row-major source). p = &src[row(lane&15)][k_base].
// lane holds row m=lane&15; K halves: e<8 -> k=e+8g ; e>=8 -> k=16+(e-8)+8g
static __device__ __forceinline__ v16h load_frag_a(const half_t* p, int g) {
    v8h lo = *(const v8h*)(p + 8 * g);
    v8h hi = *(const v8h*)(p + 16 + 8 * g);
    v16h r;
#pragma unroll
    for (int i = 0; i < 8; ++i) { r[i] = lo[i]; r[i + 8] = hi[i]; }
    return r;
}

// B fragment (32x16) from K-major storage Bt[n][k]. p = &bt[col(lane&15)][k_base].
// lane holds col n=lane&15; K halves: e<8 -> k=e+8g ; e>=8 -> k=16+(e-8)+8g
static __device__ __forceinline__ v16h load_frag_b(const half_t* p, int g) {
    v8h lo = *(const v8h*)(p + 8 * g);
    v8h hi = *(const v8h*)(p + 16 + 8 * g);
    v16h r;
#pragma unroll
    for (int i = 0; i < 8; ++i) { r[i] = lo[i]; r[i + 8] = hi[i]; }
    return r;
}

// ---------------- Kernel 1: tiled fp32 transpose -> f16 ---------------------
// in: [batches][R][Cc] fp32 ; out: [batches][Cc][R] f16. R,Cc multiples of 32.
__global__ __launch_bounds__(256)
void transpose_to_half(const float* __restrict__ in, half_t* __restrict__ out,
                       int R, int Cc) {
    __shared__ float tile[32][33];
    const int c0 = blockIdx.x * 32;
    const int r0 = blockIdx.y * 32;
    const size_t bi = (size_t)blockIdx.z * R * Cc;
    const int tx = threadIdx.x, ty = threadIdx.y;
#pragma unroll
    for (int i = 0; i < 4; ++i)
        tile[ty + i * 8][tx] = in[bi + (size_t)(r0 + ty + i * 8) * Cc + c0 + tx];
    __syncthreads();
#pragma unroll
    for (int i = 0; i < 4; ++i)
        out[bi + (size_t)(c0 + ty + i * 8) * R + r0 + tx] =
            (half_t)tile[tx][ty + i * 8];
}

// ---------------- Kernel 2: QKV GEMM (tokens @ w_qkv) + head scatter --------
// tokens: [B][N][C] f16 ; wqkvt: [3C][C] f16 (K-major for B frags)
// qh/kh: [B][H][N][D] f16 ; vt: [B][H][D][N] f16 (transposed for PV B frags)
__global__ __launch_bounds__(256)
void qkv_gemm(const half_t* __restrict__ tokens, const half_t* __restrict__ wqkvt,
              half_t* __restrict__ qh, half_t* __restrict__ kh,
              half_t* __restrict__ vt) {
    const int lane = threadIdx.x & 31;
    const int n = lane & 15, g = lane >> 4;
    const int wid = blockIdx.x * 8 + (threadIdx.x >> 5);   // 0..6143
    const int b = wid / 3072;
    const int rem = wid % 3072;
    const int m0 = (rem / 24) * 32;
    const int j0 = (rem % 24) * 64;

    const half_t* abase = tokens + (size_t)b * NTOK * CDIM;
    v8f acc[2][4];
#pragma unroll
    for (int i = 0; i < 2; ++i)
#pragma unroll
        for (int j = 0; j < 4; ++j) acc[i][j] = (v8f)(0.0f);

    const half_t* a0p = abase + (size_t)(m0 + n) * CDIM;        // n == lane&15
    const half_t* a1p = abase + (size_t)(m0 + 16 + n) * CDIM;

    for (int k0 = 0; k0 < CDIM; k0 += 32) {
        v16h a0 = load_frag_a(a0p + k0, g);
        v16h a1 = load_frag_a(a1p + k0, g);
#pragma unroll
        for (int j = 0; j < 4; ++j) {
            v16h bf = load_frag_b(wqkvt + (size_t)(j0 + j * 16 + n) * CDIM + k0, g);
            acc[0][j] = wmma16(a0, bf, acc[0][j]);
            acc[1][j] = wmma16(a1, bf, acc[1][j]);
        }
    }
    // scatter into q / k / v^T with head split: j = s*512 + h*64 + d
#pragma unroll
    for (int mf = 0; mf < 2; ++mf) {
#pragma unroll
        for (int js = 0; js < 4; ++js) {
            const int j = j0 + js * 16 + n;
            const int s = j >> 9;
            const int r = j & 511;
            const int h = r >> 6;
            const int d = r & 63;
#pragma unroll
            for (int v = 0; v < 8; ++v) {
                const int m = m0 + mf * 16 + v + 8 * g;
                const half_t val = (half_t)acc[mf][js][v];
                if (s == 0)
                    qh[(((size_t)b * HEADS + h) * NTOK + m) * DHEAD + d] = val;
                else if (s == 1)
                    kh[(((size_t)b * HEADS + h) * NTOK + m) * DHEAD + d] = val;
                else
                    vt[(((size_t)b * HEADS + h) * DHEAD + d) * NTOK + m] = val;
            }
        }
    }
}

// ---------------- Kernel 3: flash attention -----------------------------
// 8 waves/block, all sharing one (b,h); K/V chunks staged in LDS once per
// block (8x global-traffic cut), fragment reads are conflict-free ds_load_b128.
__global__ __launch_bounds__(256)
void attn_kernel(const half_t* __restrict__ qh, const half_t* __restrict__ kh,
                 const half_t* __restrict__ vt, half_t* __restrict__ attn_out) {
    const int tid = threadIdx.x;
    const int lane = tid & 31, wv = tid >> 5;
    const int n = lane & 15, g = lane >> 4;
    const int bh = blockIdx.x >> 4;                 // 0..15 = b*H + h
    const int qb = ((blockIdx.x & 15) * 8 + wv) * 32;

    __shared__ half_t kbuf[32][KPITCH];   // [key in chunk][d]
    __shared__ half_t vbuf[64][VPITCH];   // [d][key in chunk] (from v^T)
    __shared__ half_t plds[8][16 * PPITCH];
    half_t* pbuf = plds[wv];

    const half_t* qbase = qh + (size_t)bh * NTOK * DHEAD;
    const half_t* kbase = kh + (size_t)bh * NTOK * DHEAD;
    const half_t* vbase = vt + (size_t)bh * DHEAD * NTOK;

    // Q fragments, pre-scaled by D^-0.5 = 0.125 (exact in f16)
    v16h qa[2][2];
#pragma unroll
    for (int mt = 0; mt < 2; ++mt) {
        const half_t* qrow = qbase + (size_t)(qb + mt * 16 + n) * DHEAD;
        qa[mt][0] = load_frag_a(qrow, g);
        qa[mt][1] = load_frag_a(qrow + 32, g);
#pragma unroll
        for (int i = 0; i < 16; ++i) {
            qa[mt][0][i] = qa[mt][0][i] * (_Float16)0.125f;
            qa[mt][1][i] = qa[mt][1][i] * (_Float16)0.125f;
        }
    }

    v8f o[2][4];
    float mrun[2][8], lrun[2][8];
#pragma unroll
    for (int mt = 0; mt < 2; ++mt) {
#pragma unroll
        for (int dt = 0; dt < 4; ++dt) o[mt][dt] = (v8f)(0.0f);
#pragma unroll
        for (int v = 0; v < 8; ++v) { mrun[mt][v] = -3.0e38f; lrun[mt][v] = 0.0f; }
    }

    // staging assignment (256 threads move 4KB K + 4KB V per chunk)
    const int krow = tid >> 3, kseg = (tid & 7) * 8;   // K: 32 rows x 8 segs
    const int vrow = tid >> 2, vseg = (tid & 3) * 8;   // V: 64 rows x 4 segs
    const half_t* kg = kbase + (size_t)krow * DHEAD + kseg;
    const half_t* vg = vbase + (size_t)vrow * NTOK + vseg;

    for (int kc = 0; kc < NTOK; kc += 32) {
        __syncthreads();   // previous chunk fully consumed
        *(v8h*)&kbuf[krow][kseg] = *(const v8h*)(kg + (size_t)kc * DHEAD);
        *(v8h*)&vbuf[vrow][vseg] = *(const v8h*)(vg + kc);
        if (kc + 32 < NTOK) {   // prefetch next chunk (global_prefetch_b8)
            __builtin_prefetch(kg + (size_t)(kc + 32) * DHEAD, 0, 0);
            __builtin_prefetch(vg + kc + 32, 0, 0);
        }
        __syncthreads();   // staged data visible

        // K fragments: [key sub-tile][d half] ; V fragments: one per d-tile
        v16h bk[2][2];
#pragma unroll
        for (int ks = 0; ks < 2; ++ks) {
            bk[ks][0] = load_frag_b(&kbuf[ks * 16 + n][0], g);
            bk[ks][1] = load_frag_b(&kbuf[ks * 16 + n][32], g);
        }
        v16h bv[4];
#pragma unroll
        for (int dt = 0; dt < 4; ++dt)
            bv[dt] = load_frag_b(&vbuf[dt * 16 + n][0], g);

#pragma unroll
        for (int mt = 0; mt < 2; ++mt) {
            v8f s0 = (v8f)(0.0f), s1 = (v8f)(0.0f);
            s0 = wmma16(qa[mt][0], bk[0][0], s0);
            s0 = wmma16(qa[mt][1], bk[0][1], s0);
            s1 = wmma16(qa[mt][0], bk[1][0], s1);
            s1 = wmma16(qa[mt][1], bk[1][1], s1);

            float p0[8], p1[8];
#pragma unroll
            for (int v = 0; v < 8; ++v) {
                float a0 = s0[v], a1 = s1[v];          // scale folded into Q
                float cm = fmaxf(a0, a1);
                cm = fmaxf(cm, __shfl_xor(cm, 1, 32));
                cm = fmaxf(cm, __shfl_xor(cm, 2, 32));
                cm = fmaxf(cm, __shfl_xor(cm, 4, 32));
                cm = fmaxf(cm, __shfl_xor(cm, 8, 32));
                const float mold = mrun[mt][v];
                const float mnew = fmaxf(mold, cm);
                const float alpha = __expf(mold - mnew);
                const float e0 = __expf(a0 - mnew);
                const float e1 = __expf(a1 - mnew);
                float rs = e0 + e1;
                rs += __shfl_xor(rs, 1, 32);
                rs += __shfl_xor(rs, 2, 32);
                rs += __shfl_xor(rs, 4, 32);
                rs += __shfl_xor(rs, 8, 32);
                lrun[mt][v] = lrun[mt][v] * alpha + rs;
                mrun[mt][v] = mnew;
                o[mt][0][v] *= alpha;
                o[mt][1][v] *= alpha;
                o[mt][2][v] *= alpha;
                o[mt][3][v] *= alpha;
                p0[v] = e0; p1[v] = e1;
            }
            // C-layout -> A-layout re-tile of P through wave-private LDS
            // (DS ops are in-order within a wave; no barrier needed)
#pragma unroll
            for (int v = 0; v < 8; ++v) {
                const int row = v + 8 * g;
                pbuf[row * PPITCH + n] = (half_t)p0[v];
                pbuf[row * PPITCH + 16 + n] = (half_t)p1[v];
            }
            v16h pa = load_frag_a(pbuf + n * PPITCH, g);   // n == lane&15
#pragma unroll
            for (int dt = 0; dt < 4; ++dt)
                o[mt][dt] = wmma16(pa, bv[dt], o[mt][dt]);
        }
    }

    // normalize and store: attn_out[b][q][h*64 + d], f16
    const int bq = bh >> 3, hh = bh & 7;
#pragma unroll
    for (int mt = 0; mt < 2; ++mt) {
#pragma unroll
        for (int dt = 0; dt < 4; ++dt) {
#pragma unroll
            for (int v = 0; v < 8; ++v) {
                const float val = o[mt][dt][v] / lrun[mt][v];
                const int qrow = qb + mt * 16 + v + 8 * g;
                attn_out[((size_t)bq * NTOK + qrow) * CDIM + hh * DHEAD +
                         dt * 16 + n] = (half_t)val;
            }
        }
    }
}

// ---------------- Kernel 4: output projection + bias + transpose -----------
// attn: [B][N][C] f16 ; woutt: [C][C] f16 (K-major) ; out fp32 [B][C][N]
__global__ __launch_bounds__(256)
void out_proj(const half_t* __restrict__ attn, const half_t* __restrict__ woutt,
              const float* __restrict__ bout, float* __restrict__ out) {
    const int lane = threadIdx.x & 31;
    const int n = lane & 15, g = lane >> 4;
    const int wid = blockIdx.x * 8 + (threadIdx.x >> 5);   // 0..2047
    const int b = wid / 1024;
    const int rem = wid % 1024;
    const int m0 = (rem / 8) * 32;
    const int j0 = (rem % 8) * 64;

    const half_t* abase = attn + (size_t)b * NTOK * CDIM;
    const half_t* a0p = abase + (size_t)(m0 + n) * CDIM;
    const half_t* a1p = abase + (size_t)(m0 + 16 + n) * CDIM;

    v8f acc[2][4];
#pragma unroll
    for (int i = 0; i < 2; ++i)
#pragma unroll
        for (int j = 0; j < 4; ++j) acc[i][j] = (v8f)(0.0f);

    for (int k0 = 0; k0 < CDIM; k0 += 32) {
        v16h a0 = load_frag_a(a0p + k0, g);
        v16h a1 = load_frag_a(a1p + k0, g);
#pragma unroll
        for (int j = 0; j < 4; ++j) {
            v16h bf = load_frag_b(woutt + (size_t)(j0 + j * 16 + n) * CDIM + k0, g);
            acc[0][j] = wmma16(a0, bf, acc[0][j]);
            acc[1][j] = wmma16(a1, bf, acc[1][j]);
        }
    }
    // v = 0..7 are 8 consecutive output floats for a lane -> vector store
#pragma unroll
    for (int mf = 0; mf < 2; ++mf) {
#pragma unroll
        for (int js = 0; js < 4; ++js) {
            const int j = j0 + js * 16 + n;
            const float bias = bout[j];
            v8f r;
#pragma unroll
            for (int v = 0; v < 8; ++v) r[v] = acc[mf][js][v] + bias;
            const int m = m0 + mf * 16 + 8 * g;   // 32B-aligned base
            *(v8f*)&out[((size_t)b * CDIM + j) * NTOK + m] = r;
        }
    }
}

// ---------------- launch ----------------------------------------------------

extern "C" void kernel_launch(void* const* d_in, const int* in_sizes, int n_in,
                              void* d_out, int out_size, void* d_ws, size_t ws_size,
                              hipStream_t stream) {
    const float* x     = (const float*)d_in[0];   // [B,C,16,16,16] = [B,C,N]
    const float* w_qkv = (const float*)d_in[1];   // [C, 3C]
    const float* w_out = (const float*)d_in[2];   // [C, C]
    const float* b_out = (const float*)d_in[3];   // [C]
    float* out = (float*)d_out;                   // [B,C,N] fp32

    char* ws = (char*)d_ws;
    size_t off = 0;
    half_t* tokens = (half_t*)(ws + off); off += (size_t)2 * NTOK * CDIM * 2;   // 8 MB
    half_t* wqkvt  = (half_t*)(ws + off); off += (size_t)C3 * CDIM * 2;         // 1.5 MB
    half_t* woutt  = (half_t*)(ws + off); off += (size_t)CDIM * CDIM * 2;       // 0.5 MB
    half_t* qh     = (half_t*)(ws + off); off += (size_t)2 * HEADS * NTOK * DHEAD * 2;
    half_t* kh     = (half_t*)(ws + off); off += (size_t)2 * HEADS * NTOK * DHEAD * 2;
    half_t* vt     = (half_t*)(ws + off); off += (size_t)2 * HEADS * DHEAD * NTOK * 2;
    half_t* attn   = (half_t*)(ws + off); off += (size_t)2 * NTOK * CDIM * 2;
    (void)ws_size; (void)in_sizes; (void)n_in; (void)out_size;

    dim3 tb(32, 8);
    // x [B][C][N] -> tokens [B][N][C] (f16)
    transpose_to_half<<<dim3(NTOK / 32, CDIM / 32, 2), tb, 0, stream>>>(
        x, tokens, CDIM, NTOK);
    // w_qkv [C][3C] -> wqkvt [3C][C]
    transpose_to_half<<<dim3(C3 / 32, CDIM / 32, 1), tb, 0, stream>>>(
        w_qkv, wqkvt, CDIM, C3);
    // w_out [C][C] -> woutt [C][C]
    transpose_to_half<<<dim3(CDIM / 32, CDIM / 32, 1), tb, 0, stream>>>(
        w_out, woutt, CDIM, CDIM);

    // QKV projection: 6144 waves, 8 waves/block
    qkv_gemm<<<768, 256, 0, stream>>>(tokens, wqkvt, qh, kh, vt);

    // flash attention: 2048 waves, 8 waves/block sharing one (b,h)
    attn_kernel<<<256, 256, 0, stream>>>(qh, kh, vt, attn);

    // output projection: 2048 waves, 8 waves/block
    out_proj<<<256, 256, 0, stream>>>(attn, woutt, b_out, out);
}
